// KmeansCosineObjectTokens_1013612282357
// MI455X (gfx1250) — matrix operand order repaired and checked by reading, not compiled, not run on hardware
//
#include <hip/hip_runtime.h>
#include <math.h>

#define BB 16
#define NN 4096
#define CC 256
#define KK 64
#define KM_ITERS 20

typedef __attribute__((ext_vector_type(16))) __bf16 v16bf;
typedef __attribute__((ext_vector_type(8)))  float  v8f;

union FragAB { v16bf v; uint4 q[2]; };

__device__ __forceinline__ unsigned short f2bf(float f) {
  unsigned int u = __float_as_uint(f);
  u += 0x7FFFu + ((u >> 16) & 1u);     // round-to-nearest-even
  return (unsigned short)(u >> 16);
}

// ---------------- kernel: L2-normalize rows of x; emit bf16 copy + 1/norm ----------------
__global__ void __launch_bounds__(256) prep_kernel(const float* __restrict__ x,
                                                   unsigned short* __restrict__ xbf,
                                                   float* __restrict__ invnorm) {
  const int tid = threadIdx.x, wid = tid >> 5, lane = tid & 31;
  const long row = (long)blockIdx.x * 8 + wid;            // [0, B*N)
  const float* xr = x + row * CC + lane * 8;
  float4 p0 = *(const float4*)(xr);
  float4 p1 = *(const float4*)(xr + 4);
  float v[8] = {p0.x, p0.y, p0.z, p0.w, p1.x, p1.y, p1.z, p1.w};
  float ss = 0.f;
#pragma unroll
  for (int j = 0; j < 8; ++j) ss += v[j] * v[j];
#pragma unroll
  for (int m = 1; m < 32; m <<= 1) ss += __shfl_xor(ss, m, 32);
  const float inv = 1.0f / fmaxf(sqrtf(ss), 1e-12f);
  if (lane == 0) invnorm[row] = inv;
  uint4 o;
  o.x = (unsigned)f2bf(v[0] * inv) | ((unsigned)f2bf(v[1] * inv) << 16);
  o.y = (unsigned)f2bf(v[2] * inv) | ((unsigned)f2bf(v[3] * inv) << 16);
  o.z = (unsigned)f2bf(v[4] * inv) | ((unsigned)f2bf(v[5] * inv) << 16);
  o.w = (unsigned)f2bf(v[6] * inv) | ((unsigned)f2bf(v[7] * inv) << 16);
  *(uint4*)(xbf + row * CC + lane * 8) = o;
}

// ---------------- kernel: gather initial centroids (normalized rows of x) ----------------
__global__ void __launch_bounds__(256) init_kernel(const float* __restrict__ x,
                                                   const int* __restrict__ init_idx,
                                                   const float* __restrict__ invnorm,
                                                   float* __restrict__ cent,
                                                   unsigned short* __restrict__ cbf,
                                                   int* done, unsigned int* errBits) {
  const int tid = threadIdx.x, wid = tid >> 5, lane = tid & 31;
  const int gk = blockIdx.x * 8 + wid;                    // [0, B*K)
  const int b = gk >> 6, k = gk & (KK - 1);
  const int n = init_idx[b * KK + k];
  const float inv = invnorm[b * NN + n];
  const float* xr = x + ((size_t)b * NN + n) * CC + lane * 8;
  float4 p0 = *(const float4*)xr;
  float4 p1 = *(const float4*)(xr + 4);
  float w[8] = {p0.x * inv, p0.y * inv, p0.z * inv, p0.w * inv,
                p1.x * inv, p1.y * inv, p1.z * inv, p1.w * inv};
  float* cp = cent + (size_t)gk * CC + lane * 8;
  *(float4*)(cp)     = make_float4(w[0], w[1], w[2], w[3]);
  *(float4*)(cp + 4) = make_float4(w[4], w[5], w[6], w[7]);
  uint4 o;
  o.x = (unsigned)f2bf(w[0]) | ((unsigned)f2bf(w[1]) << 16);
  o.y = (unsigned)f2bf(w[2]) | ((unsigned)f2bf(w[3]) << 16);
  o.z = (unsigned)f2bf(w[4]) | ((unsigned)f2bf(w[5]) << 16);
  o.w = (unsigned)f2bf(w[6]) | ((unsigned)f2bf(w[7]) << 16);
  *(uint4*)(cbf + (size_t)gk * CC + lane * 8) = o;
  if (blockIdx.x == 0 && tid == 0) { *done = 0; *errBits = 0u; }
}

// ---------------- kernel: zero per-iteration accumulators ----------------
__global__ void __launch_bounds__(256) zero_kernel(float* __restrict__ gsums,
                                                   float* __restrict__ gcounts,
                                                   unsigned int* errBits) {
  const int i = blockIdx.x * 256 + threadIdx.x;           // grid covers B*K*C exactly
  gsums[i] = 0.f;
  if (i < BB * KK) gcounts[i] = 0.f;
  if (i == 0) *errBits = 0u;
}

// ---------------- kernel: WMMA cosine similarity + argmax + LDS-binned segment sums ------
#define CHUNKS 8
#define WAVES  8
__global__ void __launch_bounds__(256) assign_kernel(
    const float* __restrict__ x, const unsigned short* __restrict__ xbf,
    const float* __restrict__ invnorm, const unsigned short* __restrict__ cbf,
    float* __restrict__ gsums, float* __restrict__ gcounts) {
  extern __shared__ char smem[];
  float*          sSum  = (float*)smem;                                        // K*C f32
  unsigned short* sCent = (unsigned short*)(smem + KK * CC * sizeof(float));   // K*C bf16
  float*          sCnt  = (float*)(smem + KK * CC * (sizeof(float) + sizeof(unsigned short)));

  const int b = blockIdx.y;
  const int tid = threadIdx.x;

  for (int i = tid; i < KK * CC; i += 256) sSum[i] = 0.f;
  if (tid < KK) sCnt[tid] = 0.f;
  {   // stage this batch's centroids (bf16, 32KB) into LDS
    const uint4* src = (const uint4*)(cbf + (size_t)b * KK * CC);
    uint4* dst = (uint4*)sCent;
    for (int i = tid; i < KK * CC / 8; i += 256) dst[i] = src[i];
  }
  __syncthreads();

  const int lane = tid & 31, wid = tid >> 5;
  const int mrow = lane & 15, half = lane >> 4;
  const int base = blockIdx.x * (NN / CHUNKS);            // 512 rows per block

  for (int pass = 0; pass < (NN / CHUNKS) / (WAVES * 16); ++pass) {  // 4 passes
    const int n0 = base + pass * (WAVES * 16) + wid * 16;
    v8f acc[4] = {};
    // A fragment: lane holds row (n0+mrow); VGPR0-3 = K 0..7 (+16 for 4-7), halves split 8..15 / 24..31
    const unsigned short* arow = xbf + ((size_t)b * NN + n0 + mrow) * CC;
#pragma unroll
    for (int c0 = 0; c0 < CC; c0 += 32) {
      FragAB a;
      a.q[0] = *(const uint4*)(arow + c0 + half * 8);
      a.q[1] = *(const uint4*)(arow + c0 + 16 + half * 8);
#pragma unroll
      for (int t = 0; t < 4; ++t) {
        // B fragment: lane = centroid column (t*16+mrow); VGPR0-3 = K 0..7 (+16 for 4-7), halves split 8..15 / 24..31
        FragAB bm;
        const unsigned short* brow = sCent + (size_t)(t * 16 + mrow) * CC + c0 + half * 8;
        bm.q[0] = *(const uint4*)(brow);
        bm.q[1] = *(const uint4*)(brow + 16);
        acc[t] = __builtin_amdgcn_wmma_f32_16x16x32_bf16(false, a.v, false, bm.v,
                                                         (short)0, acc[t], false, false);
      }
    }
    // D layout: acc[t][r] in lane L holds cos(row = (L<16 ? r : r+8), col = t*16 + L%16)
#pragma unroll
    for (int r = 0; r < 8; ++r) {
      float bv = acc[0][r]; int bi = mrow;
#pragma unroll
      for (int t = 1; t < 4; ++t) {
        const float cv = acc[t][r]; const int ci = t * 16 + mrow;
        if (cv > bv) { bv = cv; bi = ci; }                // strict > keeps first max (low index)
      }
#pragma unroll
      for (int m = 1; m < 16; m <<= 1) {                  // butterfly within each 16-lane half
        const float ov = __shfl_xor(bv, m, 32);
        const int   oi = __shfl_xor(bi, m, 32);
        if (ov > bv || (ov == bv && oi < bi)) { bv = ov; bi = oi; }
      }
      const int kLo = __shfl(bi, 0, 32);                  // argmax of row n0+r
      const int kHi = __shfl(bi, 16, 32);                 // argmax of row n0+r+8
      {
        const int row = n0 + r;
        const float inv = invnorm[b * NN + row];
        const float* xr = x + ((size_t)b * NN + row) * CC + lane * 8;
        float4 p0 = *(const float4*)xr;
        float4 p1 = *(const float4*)(xr + 4);
        float* sd = sSum + kLo * CC + lane * 8;
        atomicAdd(&sd[0], p0.x * inv); atomicAdd(&sd[1], p0.y * inv);
        atomicAdd(&sd[2], p0.z * inv); atomicAdd(&sd[3], p0.w * inv);
        atomicAdd(&sd[4], p1.x * inv); atomicAdd(&sd[5], p1.y * inv);
        atomicAdd(&sd[6], p1.z * inv); atomicAdd(&sd[7], p1.w * inv);
        if (lane == 0) atomicAdd(&sCnt[kLo], 1.0f);
      }
      {
        const int row = n0 + r + 8;
        const float inv = invnorm[b * NN + row];
        const float* xr = x + ((size_t)b * NN + row) * CC + lane * 8;
        float4 p0 = *(const float4*)xr;
        float4 p1 = *(const float4*)(xr + 4);
        float* sd = sSum + kHi * CC + lane * 8;
        atomicAdd(&sd[0], p0.x * inv); atomicAdd(&sd[1], p0.y * inv);
        atomicAdd(&sd[2], p0.z * inv); atomicAdd(&sd[3], p0.w * inv);
        atomicAdd(&sd[4], p1.x * inv); atomicAdd(&sd[5], p1.y * inv);
        atomicAdd(&sd[6], p1.z * inv); atomicAdd(&sd[7], p1.w * inv);
        if (lane == 16) atomicAdd(&sCnt[kHi], 1.0f);
      }
    }
  }
  __syncthreads();
  float* gs = gsums + (size_t)b * KK * CC;
  for (int i = tid; i < KK * CC; i += 256) atomicAdd(&gs[i], sSum[i]);
  if (tid < KK) atomicAdd(&gcounts[b * KK + tid], sCnt[tid]);
}

// ---------------- kernel: mean -> renormalize -> error max -> freeze-aware write ---------
__global__ void __launch_bounds__(256) update_kernel(
    const float* __restrict__ gsums, const float* __restrict__ gcounts,
    float* __restrict__ cent, unsigned short* __restrict__ cbf,
    unsigned int* errBits, const int* __restrict__ done) {
  const int tid = threadIdx.x, wid = tid >> 5, lane = tid & 31;
  const int gk = blockIdx.x * 8 + wid;                    // [0, B*K)
  const float cnt = fmaxf(gcounts[gk], 1.0f);
  const float* sp = gsums + (size_t)gk * CC + lane * 8;
  float4 p0 = *(const float4*)sp;
  float4 p1 = *(const float4*)(sp + 4);
  float v[8] = {p0.x, p0.y, p0.z, p0.w, p1.x, p1.y, p1.z, p1.w};
  float ss = 0.f;
#pragma unroll
  for (int j = 0; j < 8; ++j) { v[j] /= cnt; ss += v[j] * v[j]; }
#pragma unroll
  for (int m = 1; m < 32; m <<= 1) ss += __shfl_xor(ss, m, 32);
  const float inv = 1.0f / fmaxf(sqrtf(ss), 1e-20f);
  const int dn = *done;                                   // freeze uses pre-update done
  float* cp = cent + (size_t)gk * CC + lane * 8;
  unsigned short* cb = cbf + (size_t)gk * CC + lane * 8;
  float emax = 0.f, outv[8];
#pragma unroll
  for (int j = 0; j < 8; ++j) {
    const float nc = v[j] * inv;
    const float old = cp[j];
    emax = fmaxf(emax, fabsf(old - nc));
    outv[j] = dn ? old : nc;
  }
  *(float4*)(cp)     = make_float4(outv[0], outv[1], outv[2], outv[3]);
  *(float4*)(cp + 4) = make_float4(outv[4], outv[5], outv[6], outv[7]);
  uint4 o;
  o.x = (unsigned)f2bf(outv[0]) | ((unsigned)f2bf(outv[1]) << 16);
  o.y = (unsigned)f2bf(outv[2]) | ((unsigned)f2bf(outv[3]) << 16);
  o.z = (unsigned)f2bf(outv[4]) | ((unsigned)f2bf(outv[5]) << 16);
  o.w = (unsigned)f2bf(outv[6]) | ((unsigned)f2bf(outv[7]) << 16);
  *(uint4*)cb = o;
#pragma unroll
  for (int m = 1; m < 32; m <<= 1) emax = fmaxf(emax, __shfl_xor(emax, m, 32));
  if (lane == 0) atomicMax(errBits, __float_as_uint(emax)); // non-negative floats order as uints
}

__global__ void done_kernel(const unsigned int* errBits, int* done) {
  if (__uint_as_float(*errBits) < 1e-6f) *done = 1;
}

extern "C" void kernel_launch(void* const* d_in, const int* in_sizes, int n_in,
                              void* d_out, int out_size, void* d_ws, size_t ws_size,
                              hipStream_t stream) {
  (void)in_sizes; (void)n_in; (void)out_size; (void)ws_size;
  const float* x        = (const float*)d_in[0];          // [B,N,C] fp32
  const int*   init_idx = (const int*)d_in[1];            // [B,K] int32
  float*       cent     = (float*)d_out;                  // [B,K,C] fp32 output = centroid buffer

  char* ws = (char*)d_ws;
  size_t off = 0;
  auto alloc = [&](size_t bytes) -> char* {
    off = (off + 255) & ~(size_t)255;
    char* p = ws + off;
    off += bytes;
    return p;
  };
  unsigned short* xbf     = (unsigned short*)alloc((size_t)BB * NN * CC * 2);  // 32 MB
  float*          invnorm = (float*)alloc((size_t)BB * NN * 4);
  unsigned short* cbf     = (unsigned short*)alloc((size_t)BB * KK * CC * 2);
  float*          gsums   = (float*)alloc((size_t)BB * KK * CC * 4);
  float*          gcounts = (float*)alloc((size_t)BB * KK * 4);
  unsigned int*   errBits = (unsigned int*)alloc(4);
  int*            done    = (int*)alloc(4);

  prep_kernel<<<BB * NN / 8, 256, 0, stream>>>(x, xbf, invnorm);
  init_kernel<<<BB * KK / 8, 256, 0, stream>>>(x, init_idx, invnorm, cent, cbf, done, errBits);

  const size_t smem = (size_t)KK * CC * 4 + (size_t)KK * CC * 2 + (size_t)KK * 4; // ~96.25 KB
  for (int it = 0; it < KM_ITERS; ++it) {
    zero_kernel<<<BB * KK * CC / 256, 256, 0, stream>>>(gsums, gcounts, errBits);
    assign_kernel<<<dim3(CHUNKS, BB), 256, smem, stream>>>(x, xbf, invnorm, cbf, gsums, gcounts);
    update_kernel<<<BB * KK / 8, 256, 0, stream>>>(gsums, gcounts, cent, cbf, errBits, done);
    done_kernel<<<1, 1, 0, stream>>>(errBits, done);
  }
}
